// Equi_Nonlin_Grad_Module_71554155151850
// MI455X (gfx1250) — matrix operand order repaired and checked by reading, not compiled, not run on hardware
//
#include <hip/hip_runtime.h>
#include <hip/hip_bf16.h>
#include <cmath>

typedef __attribute__((ext_vector_type(16))) __bf16 v16bf;
typedef __attribute__((ext_vector_type(8)))  float  v8f;

#define DIM0 128
#define DIMT 480

// workspace offsets (bytes)
#define OFF_WCAT 0
#define SZ_WCAT  (672*4*512*2)
#define OFF_WB0  (OFF_WCAT + SZ_WCAT)
#define SZ_WB0   (256*8*512*2)
#define OFF_WB1  (OFF_WB0 + SZ_WB0)
#define SZ_WB1   (128*4*512*2)
#define OFF_WB2  (OFF_WB1 + SZ_WB1)
#define SZ_WB2   (64*2*512*2)
#define OFF_W1P  (OFF_WB2 + SZ_WB2)
#define OFF_W2P  (OFF_W1P + 8192)
#define OFF_W3P  (OFF_W2P + 8192)
#define OFF_W1T  (OFF_W3P + 8192)
#define OFF_W2T  (OFF_W1T + 8192)
#define OFF_C3   (OFF_W2T + 8192)

// LDS layout
#define IS 488   // padded input row stride (floats), keeps 16B row alignment
#define TS 72    // padded tile row stride
#define LDS_FLOATS (16*IS + 5*16*TS)

// ---------------- weight pre-pack kernels (B-fragment layout) ----------------
// B-fragment element mapping: k = c*32 + (lane>>4)*8 + e + (e>>3)*8 ; col = j*16 + (lane&15)

__global__ void pack_fwd_kernel(const float* __restrict__ W0, const float* __restrict__ W1,
                                const float* __restrict__ W2, __bf16* __restrict__ out,
                                float s0, float s1, float s2) {
  int tid = blockIdx.x * blockDim.x + threadIdx.x;
  if (tid >= 672 * 2048) return;
  int e = tid & 15, lane = (tid >> 4) & 31, j = (tid >> 9) & 3, c = tid >> 11;
  int k = c * 32 + ((lane >> 4) << 3) + e + ((e >> 3) << 3);
  int col = (j << 4) + (lane & 15);
  float v;
  if (k < 16384)      { int u = k >> 7, vv = k & 127;                     v = s0 * W0[((u << 7) + vv) * 64 + col]; }
  else if (k < 20480) { int kk = k - 16384; int u = kk >> 6, vv = kk & 63; v = s1 * W1[((u << 6) + vv) * 64 + col]; }
  else                { int kk = k - 20480; int u = kk >> 5, vv = kk & 31; v = s2 * W2[((u << 5) + vv) * 64 + col]; }
  out[tid] = (__bf16)v;
}

__global__ void pack_bwd_kernel(const float* __restrict__ W, __bf16* __restrict__ out,
                                int mode, float s, int total) {
  int tid = blockIdx.x * blockDim.x + threadIdx.x;
  if (tid >= total) return;
  int e = tid & 15, lane = (tid >> 4) & 31;
  int j, c, M;
  if (mode == 0)      { j = (tid >> 9) & 7; c = tid >> 12; M = 128; }
  else if (mode == 1) { j = (tid >> 9) & 3; c = tid >> 11; M = 64; }
  else                { j = (tid >> 9) & 1; c = tid >> 10; M = 32; }
  int k = c * 32 + ((lane >> 4) << 3) + e + ((e >> 3) << 3);   // k = vv*64 + h
  int vv = k >> 6, h = k & 63;
  int u = (j << 4) + (lane & 15);
  float v = s * (W[(u * M + vv) * 64 + h] + W[(vv * M + u) * 64 + h]);
  out[tid] = (__bf16)v;
}

__global__ void pack_mlp_kernel(const float* __restrict__ w1, const float* __restrict__ w2,
                                const float* __restrict__ w3, __bf16* __restrict__ out) {
  int tid = blockIdx.x * blockDim.x + threadIdx.x;
  if (tid >= 5 * 4096) return;
  int which = tid >> 12, t = tid & 4095;
  int e = t & 15, lane = (t >> 4) & 31, j = (t >> 9) & 3, c = t >> 11;
  int k = c * 32 + ((lane >> 4) << 3) + e + ((e >> 3) << 3);
  int col = (j << 4) + (lane & 15);
  float v = 0.f;
  switch (which) {
    case 0: v = w1[k * 64 + col]; break;     // w1
    case 1: v = w2[k * 64 + col]; break;     // w2
    case 2: v = w3[k * 64 + col]; break;     // w3
    case 3: v = w1[col * 64 + k]; break;     // w1^T
    case 4: v = w2[col * 64 + k]; break;     // w2^T
  }
  out[tid] = (__bf16)v;
}

__global__ void c3_kernel(const float* __restrict__ w3, float* __restrict__ c3) {
  int j = threadIdx.x;
  if (j < 64) {
    float s = 0.f;
    for (int z = 0; z < 64; ++z) s += w3[j * 64 + z];
    c3[j] = s;
  }
}

// ---------------- device helpers ----------------

__device__ __forceinline__ v8f wmma_bf(v16bf a, v16bf b, v8f c) {
  return __builtin_amdgcn_wmma_f32_16x16x32_bf16(false, a, false, b, (short)0, c, false, false);
}
__device__ __forceinline__ float sigm(float x) { return 1.0f / (1.0f + __expf(-x)); }
__device__ __forceinline__ __bf16 bf(float x) { return (__bf16)x; }

// A-fragment mapping per ISA: lane<16 -> K = e (e<8) / e+8 (e>=8); lane>=16 -> +8.
// strip s0..s3 carries rows {kbase+hi*8 + 0..7} and {kbase+hi*8+16 .. +23}.
__device__ __forceinline__ v16bf rank1_frag(float x, float4 s0, float4 s1, float4 s2, float4 s3) {
  v16bf a;
  a[0] = bf(x * s0.x); a[1] = bf(x * s0.y); a[2] = bf(x * s0.z); a[3] = bf(x * s0.w);
  a[4] = bf(x * s1.x); a[5] = bf(x * s1.y); a[6] = bf(x * s1.z); a[7] = bf(x * s1.w);
  a[8] = bf(x * s2.x); a[9] = bf(x * s2.y); a[10] = bf(x * s2.z); a[11] = bf(x * s2.w);
  a[12] = bf(x * s3.x); a[13] = bf(x * s3.y); a[14] = bf(x * s3.z); a[15] = bf(x * s3.w);
  return a;
}

// vectorized (float4) A fragment from a padded LDS tile; p must be 16B aligned
__device__ __forceinline__ v16bf frag_tile4(const float* p) {
  const float4* q = (const float4*)p;
  float4 f0 = q[0], f1 = q[1], f2 = q[4], f3 = q[5];
  v16bf a;
  a[0] = bf(f0.x); a[1] = bf(f0.y); a[2] = bf(f0.z); a[3] = bf(f0.w);
  a[4] = bf(f1.x); a[5] = bf(f1.y); a[6] = bf(f1.z); a[7] = bf(f1.w);
  a[8] = bf(f2.x); a[9] = bf(f2.y); a[10] = bf(f2.z); a[11] = bf(f2.w);
  a[12] = bf(f3.x); a[13] = bf(f3.y); a[14] = bf(f3.z); a[15] = bf(f3.w);
  return a;
}

// rank-1 GEMM over nv chunks, 4 output tiles, double-buffered B, peeled tail.
// A(chunk v) = xsrc[v*xstride] * strip ; B fragment block at base + v*step (v16bf units)
__device__ __forceinline__ void rank1_gemm4(const float* xsrc, int xstride,
                                            const float* stripBase,
                                            const v16bf* base, int step, int nv,
                                            v8f acc[4]) {
  const float4* q = (const float4*)stripBase;
  float4 s0 = q[0], s1 = q[1], s2 = q[4], s3 = q[5];
  const v16bf* bp = base;
  v16bf b0 = bp[0], b1 = bp[32], b2 = bp[64], b3 = bp[96];
#pragma unroll 2
  for (int v = 0; v < nv - 1; ++v) {
    v16bf a = rank1_frag(xsrc[v * xstride], s0, s1, s2, s3);
    const v16bf* bn = bp + step;
    v16bf n0 = bn[0], n1 = bn[32], n2 = bn[64], n3 = bn[96];
    acc[0] = wmma_bf(a, b0, acc[0]);
    acc[1] = wmma_bf(a, b1, acc[1]);
    acc[2] = wmma_bf(a, b2, acc[2]);
    acc[3] = wmma_bf(a, b3, acc[3]);
    b0 = n0; b1 = n1; b2 = n2; b3 = n3; bp = bn;
  }
  v16bf a = rank1_frag(xsrc[(nv - 1) * xstride], s0, s1, s2, s3);
  acc[0] = wmma_bf(a, b0, acc[0]);
  acc[1] = wmma_bf(a, b1, acc[1]);
  acc[2] = wmma_bf(a, b2, acc[2]);
  acc[3] = wmma_bf(a, b3, acc[3]);
}

// 2-tile variant (for the 32-column dx2 GEMM)
__device__ __forceinline__ void rank1_gemm2(const float* xsrc, int xstride,
                                            const float* stripBase,
                                            const v16bf* base, int step, int nv,
                                            v8f acc[2]) {
  const float4* q = (const float4*)stripBase;
  float4 s0 = q[0], s1 = q[1], s2 = q[4], s3 = q[5];
  const v16bf* bp = base;
  v16bf b0 = bp[0], b1 = bp[32];
#pragma unroll 2
  for (int v = 0; v < nv - 1; ++v) {
    v16bf a = rank1_frag(xsrc[v * xstride], s0, s1, s2, s3);
    const v16bf* bn = bp + step;
    v16bf n0 = bn[0], n1 = bn[32];
    acc[0] = wmma_bf(a, b0, acc[0]);
    acc[1] = wmma_bf(a, b1, acc[1]);
    b0 = n0; b1 = n1; bp = bn;
  }
  v16bf a = rank1_frag(xsrc[(nv - 1) * xstride], s0, s1, s2, s3);
  acc[0] = wmma_bf(a, b0, acc[0]);
  acc[1] = wmma_bf(a, b1, acc[1]);
}

__device__ __forceinline__ void gemm64(const float* tile, const v16bf* B, v8f acc[4],
                                       int row, int hi, int lane) {
#pragma unroll
  for (int c = 0; c < 2; ++c) {
    v16bf a = frag_tile4(tile + row * TS + c * 32 + hi * 8);
    const v16bf* bp = B + c * 128 + lane;
    v16bf b0 = bp[0], b1 = bp[32], b2 = bp[64], b3 = bp[96];
    acc[0] = wmma_bf(a, b0, acc[0]);
    acc[1] = wmma_bf(a, b1, acc[1]);
    acc[2] = wmma_bf(a, b2, acc[2]);
    acc[3] = wmma_bf(a, b3, acc[3]);
  }
}

// ---------------- fused forward+backward kernel: 1 wave == 16 samples ----------------

__global__ __attribute__((amdgpu_flat_work_group_size(32, 32)))
void fctp_fused_kernel(const float* __restrict__ tin,
                       const float* __restrict__ b1, const float* __restrict__ b2,
                       const float* __restrict__ b3,
                       const unsigned char* __restrict__ ws,
                       float* __restrict__ dout, int Ntot) {
  extern __shared__ float lds[];
  const int lane = threadIdx.x;
  const int row = lane & 15;
  const int hi = lane >> 4;
  const int col16 = row;
  const int n0 = blockIdx.x * 16;

  float* inT = lds;
  float* T0 = lds + 16 * IS;      // hTP, later g_a2
  float* T1 = T0 + 16 * TS;       // a1
  float* T2 = T1 + 16 * TS;       // h1, later g_a1
  float* T3 = T2 + 16 * TS;       // a2
  float* T4 = T3 + 16 * TS;       // h2, later g_hTP

  const v16bf* Wcat = (const v16bf*)(ws + OFF_WCAT);
  const v16bf* Wb0 = (const v16bf*)(ws + OFF_WB0);
  const v16bf* Wb1 = (const v16bf*)(ws + OFF_WB1);
  const v16bf* Wb2 = (const v16bf*)(ws + OFF_WB2);
  const v16bf* W1p = (const v16bf*)(ws + OFF_W1P);
  const v16bf* W2p = (const v16bf*)(ws + OFF_W2P);
  const v16bf* W3p = (const v16bf*)(ws + OFF_W3P);
  const v16bf* W1t = (const v16bf*)(ws + OFF_W1T);
  const v16bf* W2t = (const v16bf*)(ws + OFF_W2T);
  const float* c3 = (const float*)(ws + OFF_C3);

  // stage 16x480 inputs into LDS (float4, coalesced)
  {
    const float4* src = (const float4*)(tin + (size_t)n0 * DIMT);
    for (int i = lane; i < 16 * DIMT / 4; i += 32) {
      float4 v = src[i];
      int r = (i * 4) / DIMT, d = (i * 4) % DIMT;
      float* dst = inT + r * IS + d;
      dst[0] = v.x; dst[1] = v.y; dst[2] = v.z; dst[3] = v.w;
    }
  }
  __syncthreads();

  const float* xr = inT + row * IS;

  // ---- forward tensor product: hTP = A(21504) x Wcat ----
  v8f acc[4] = {};
  // region 0: x0 outer product; c = u*4 + vb, strip = x0[vb*32 + hi*8 ...]
#pragma unroll 1
  for (int vb = 0; vb < 4; ++vb)
    rank1_gemm4(xr, 1, xr + (vb << 5) + hi * 8,
                Wcat + vb * 128 + lane, 512, 128, acc);

  // region 1: x1 dot3; c = 512 + u*2 + vs
#pragma unroll 1
  for (int vs = 0; vs < 2; ++vs) {
    const float* xv = xr + DIM0 + ((vs << 5) + hi * 8) * 3;
    float t0[24], t1[24];
    const float4* q0 = (const float4*)xv;
    const float4* q1 = (const float4*)(xv + 48);
#pragma unroll
    for (int i = 0; i < 6; ++i) {
      float4 a4 = q0[i]; t0[4 * i] = a4.x; t0[4 * i + 1] = a4.y; t0[4 * i + 2] = a4.z; t0[4 * i + 3] = a4.w;
      float4 b4 = q1[i]; t1[4 * i] = b4.x; t1[4 * i + 1] = b4.y; t1[4 * i + 2] = b4.z; t1[4 * i + 3] = b4.w;
    }
    const v16bf* bp = Wcat + (size_t)(512 + vs) * 128 + lane;
    v16bf b0 = bp[0], b1v = bp[32], b2v = bp[64], b3v = bp[96];
#pragma unroll 2
    for (int u = 0; u < 63; ++u) {
      const float* xu = xr + DIM0 + u * 3;
      float u0 = xu[0], u1 = xu[1], u2 = xu[2];
      v16bf a;
#pragma unroll
      for (int e = 0; e < 8; ++e) {
        a[e]     = bf(u0 * t0[3 * e] + u1 * t0[3 * e + 1] + u2 * t0[3 * e + 2]);
        a[e + 8] = bf(u0 * t1[3 * e] + u1 * t1[3 * e + 1] + u2 * t1[3 * e + 2]);
      }
      const v16bf* bn = bp + 256;
      v16bf n0 = bn[0], n1 = bn[32], n2 = bn[64], n3 = bn[96];
      acc[0] = wmma_bf(a, b0, acc[0]);
      acc[1] = wmma_bf(a, b1v, acc[1]);
      acc[2] = wmma_bf(a, b2v, acc[2]);
      acc[3] = wmma_bf(a, b3v, acc[3]);
      b0 = n0; b1v = n1; b2v = n2; b3v = n3; bp = bn;
    }
    {
      const float* xu = xr + DIM0 + 63 * 3;
      float u0 = xu[0], u1 = xu[1], u2 = xu[2];
      v16bf a;
#pragma unroll
      for (int e = 0; e < 8; ++e) {
        a[e]     = bf(u0 * t0[3 * e] + u1 * t0[3 * e + 1] + u2 * t0[3 * e + 2]);
        a[e + 8] = bf(u0 * t1[3 * e] + u1 * t1[3 * e + 1] + u2 * t1[3 * e + 2]);
      }
      acc[0] = wmma_bf(a, b0, acc[0]);
      acc[1] = wmma_bf(a, b1v, acc[1]);
      acc[2] = wmma_bf(a, b2v, acc[2]);
      acc[3] = wmma_bf(a, b3v, acc[3]);
    }
  }

  // region 2: x2 dot5; c = 640 + u
  {
    const float* xv = xr + 320 + hi * 40;
    float t0[40], t1[40];
    const float4* q0 = (const float4*)xv;
    const float4* q1 = (const float4*)(xv + 80);
#pragma unroll
    for (int i = 0; i < 10; ++i) {
      float4 a4 = q0[i]; t0[4 * i] = a4.x; t0[4 * i + 1] = a4.y; t0[4 * i + 2] = a4.z; t0[4 * i + 3] = a4.w;
      float4 b4 = q1[i]; t1[4 * i] = b4.x; t1[4 * i + 1] = b4.y; t1[4 * i + 2] = b4.z; t1[4 * i + 3] = b4.w;
    }
    const v16bf* bp = Wcat + (size_t)640 * 128 + lane;
    v16bf b0 = bp[0], b1v = bp[32], b2v = bp[64], b3v = bp[96];
#pragma unroll 2
    for (int u = 0; u < 31; ++u) {
      const float* xu = xr + 320 + u * 5;
      float u0 = xu[0], u1 = xu[1], u2 = xu[2], u3 = xu[3], u4 = xu[4];
      v16bf a;
#pragma unroll
      for (int e = 0; e < 8; ++e) {
        a[e]     = bf(u0 * t0[5 * e] + u1 * t0[5 * e + 1] + u2 * t0[5 * e + 2] + u3 * t0[5 * e + 3] + u4 * t0[5 * e + 4]);
        a[e + 8] = bf(u0 * t1[5 * e] + u1 * t1[5 * e + 1] + u2 * t1[5 * e + 2] + u3 * t1[5 * e + 3] + u4 * t1[5 * e + 4]);
      }
      const v16bf* bn = bp + 128;
      v16bf n0 = bn[0], n1 = bn[32], n2 = bn[64], n3 = bn[96];
      acc[0] = wmma_bf(a, b0, acc[0]);
      acc[1] = wmma_bf(a, b1v, acc[1]);
      acc[2] = wmma_bf(a, b2v, acc[2]);
      acc[3] = wmma_bf(a, b3v, acc[3]);
      b0 = n0; b1v = n1; b2v = n2; b3v = n3; bp = bn;
    }
    {
      const float* xu = xr + 320 + 31 * 5;
      float u0 = xu[0], u1 = xu[1], u2 = xu[2], u3 = xu[3], u4 = xu[4];
      v16bf a;
#pragma unroll
      for (int e = 0; e < 8; ++e) {
        a[e]     = bf(u0 * t0[5 * e] + u1 * t0[5 * e + 1] + u2 * t0[5 * e + 2] + u3 * t0[5 * e + 3] + u4 * t0[5 * e + 4]);
        a[e + 8] = bf(u0 * t1[5 * e] + u1 * t1[5 * e + 1] + u2 * t1[5 * e + 2] + u3 * t1[5 * e + 3] + u4 * t1[5 * e + 4]);
      }
      acc[0] = wmma_bf(a, b0, acc[0]);
      acc[1] = wmma_bf(a, b1v, acc[1]);
      acc[2] = wmma_bf(a, b2v, acc[2]);
      acc[3] = wmma_bf(a, b3v, acc[3]);
    }
  }

#pragma unroll
  for (int j = 0; j < 4; ++j)
#pragma unroll
    for (int r = 0; r < 8; ++r)
      T0[(r + hi * 8) * TS + j * 16 + col16] = acc[j][r];
  __syncthreads();

  // ---- MLP forward ----
  v8f m1[4] = {};
  gemm64(T0, W1p, m1, row, hi, lane);
#pragma unroll
  for (int j = 0; j < 4; ++j) {
    float bb = b1[j * 16 + col16];
#pragma unroll
    for (int r = 0; r < 8; ++r) {
      int rm = r + hi * 8;
      float x = m1[j][r] + bb;
      T1[rm * TS + j * 16 + col16] = x;
      T2[rm * TS + j * 16 + col16] = x * sigm(x);
    }
  }
  __syncthreads();

  v8f m2[4] = {};
  gemm64(T2, W2p, m2, row, hi, lane);
#pragma unroll
  for (int j = 0; j < 4; ++j) {
    float bb = b2[j * 16 + col16];
#pragma unroll
    for (int r = 0; r < 8; ++r) {
      int rm = r + hi * 8;
      float x = m2[j][r] + bb;
      T3[rm * TS + j * 16 + col16] = x;
      T4[rm * TS + j * 16 + col16] = x * sigm(x);
    }
  }
  __syncthreads();

  v8f m3[4] = {};
  gemm64(T4, W3p, m3, row, hi, lane);
#pragma unroll
  for (int j = 0; j < 4; ++j) {
    float bb = b3[j * 16 + col16];
#pragma unroll
    for (int r = 0; r < 8; ++r)
      dout[(size_t)(n0 + r + hi * 8) * 64 + j * 16 + col16] = m3[j][r] + bb;
  }

  // ---- backward through MLP ----
  // g_a2 = c3 * silu'(a2)  -> T0
  for (int i = lane; i < 1024; i += 32) {
    int r2 = i >> 6, cc = i & 63;
    float a2 = T3[r2 * TS + cc];
    float s = sigm(a2);
    T0[r2 * TS + cc] = c3[cc] * (s * (1.f + a2 * (1.f - s)));
  }
  __syncthreads();

  v8f g1[4] = {};
  gemm64(T0, W2t, g1, row, hi, lane);       // g_h1
#pragma unroll
  for (int j = 0; j < 4; ++j)
#pragma unroll
    for (int r = 0; r < 8; ++r) {
      int rm = r + hi * 8, cc = j * 16 + col16;
      float a1v = T1[rm * TS + cc];
      float s = sigm(a1v);
      T2[rm * TS + cc] = g1[j][r] * (s * (1.f + a1v * (1.f - s)));   // g_a1
    }
  __syncthreads();

  v8f g0[4] = {};
  gemm64(T2, W1t, g0, row, hi, lane);       // g_hTP
#pragma unroll
  for (int j = 0; j < 4; ++j)
#pragma unroll
    for (int r = 0; r < 8; ++r)
      T4[(r + hi * 8) * TS + j * 16 + col16] = g0[j][r];
  __syncthreads();

  float* yout = dout + (size_t)Ntot * 64;
  const float* gr = T4 + row * TS;

  // ---- dx0 = (x0 (x) g) * Wb0 : K=8192, 128 cols; split 2 col-halves x 2 h-halves ----
  {
    v8f a8[8] = {};
#pragma unroll
    for (int jh = 0; jh < 2; ++jh)
#pragma unroll
      for (int hs = 0; hs < 2; ++hs)
        rank1_gemm4(xr, 1, gr + (hs << 5) + hi * 8,
                    Wb0 + (hs * 8 + jh * 4) * 32 + lane, 512, 128, a8 + jh * 4);
#pragma unroll
    for (int j = 0; j < 8; ++j)
#pragma unroll
      for (int r = 0; r < 8; ++r)
        yout[(size_t)(n0 + r + hi * 8) * DIMT + j * 16 + col16] = a8[j][r];
  }

  // ---- dx1 (m = 0..2): K=4096, 64 cols each ----
#pragma unroll 1
  for (int m = 0; m < 3; ++m) {
    v8f a4[4] = {};
#pragma unroll
    for (int hs = 0; hs < 2; ++hs)
      rank1_gemm4(xr + DIM0 + m, 3, gr + (hs << 5) + hi * 8,
                  Wb1 + hs * 128 + lane, 256, 64, a4);
#pragma unroll
    for (int j = 0; j < 4; ++j)
#pragma unroll
      for (int r = 0; r < 8; ++r)
        yout[(size_t)(n0 + r + hi * 8) * DIMT + DIM0 + (j * 16 + col16) * 3 + m] = a4[j][r];
  }

  // ---- dx2 (m = 0..4): K=2048, 32 cols each ----
#pragma unroll 1
  for (int m = 0; m < 5; ++m) {
    v8f a2v[2] = {};
#pragma unroll
    for (int hs = 0; hs < 2; ++hs)
      rank1_gemm2(xr + 320 + m, 5, gr + (hs << 5) + hi * 8,
                  Wb2 + hs * 64 + lane, 128, 32, a2v);
#pragma unroll
    for (int j = 0; j < 2; ++j)
#pragma unroll
      for (int r = 0; r < 8; ++r)
        yout[(size_t)(n0 + r + hi * 8) * DIMT + 320 + (j * 16 + col16) * 5 + m] = a2v[j][r];
  }
}

// ---------------- host launcher ----------------

extern "C" void kernel_launch(void* const* d_in, const int* in_sizes, int n_in,
                              void* d_out, int out_size, void* d_ws, size_t ws_size,
                              hipStream_t stream) {
  const float* tin = (const float*)d_in[0];
  const float* W0 = (const float*)d_in[1];
  const float* W1 = (const float*)d_in[2];
  const float* W2 = (const float*)d_in[3];
  const float* w1 = (const float*)d_in[4];
  const float* b1 = (const float*)d_in[5];
  const float* w2 = (const float*)d_in[6];
  const float* b2 = (const float*)d_in[7];
  const float* w3 = (const float*)d_in[8];
  const float* b3 = (const float*)d_in[9];
  unsigned char* ws = (unsigned char*)d_ws;
  float* out = (float*)d_out;
  int Ntot = in_sizes[0] / DIMT;

  float s0 = 1.0f / sqrtf((float)(128 * 128 + 64 * 64 + 32 * 32));
  float s1 = s0 / sqrtf(3.0f);
  float s2 = s0 / sqrtf(5.0f);

  pack_fwd_kernel<<<(672 * 2048 + 255) / 256, 256, 0, stream>>>(
      W0, W1, W2, (__bf16*)(ws + OFF_WCAT), s0, s1, s2);
  pack_bwd_kernel<<<(256 * 8 * 512 + 255) / 256, 256, 0, stream>>>(
      W0, (__bf16*)(ws + OFF_WB0), 0, s0, 256 * 8 * 512);
  pack_bwd_kernel<<<(128 * 4 * 512 + 255) / 256, 256, 0, stream>>>(
      W1, (__bf16*)(ws + OFF_WB1), 1, s1, 128 * 4 * 512);
  pack_bwd_kernel<<<(64 * 2 * 512 + 255) / 256, 256, 0, stream>>>(
      W2, (__bf16*)(ws + OFF_WB2), 2, s2, 64 * 2 * 512);
  pack_mlp_kernel<<<(5 * 4096 + 255) / 256, 256, 0, stream>>>(
      w1, w2, w3, (__bf16*)(ws + OFF_W1P));
  c3_kernel<<<1, 64, 0, stream>>>(w3, (float*)(ws + OFF_C3));

  size_t ldsBytes = LDS_FLOATS * sizeof(float);
  fctp_fused_kernel<<<Ntot / 16, 32, ldsBytes, stream>>>(tin, b1, b2, b3, ws, out, Ntot);
}
